// SinkhornDistance_89833535963740
// MI455X (gfx1250) — matrix operand order repaired and checked by reading, not compiled, not run on hardware
//
#include <hip/hip_runtime.h>
#include <math.h>

typedef __attribute__((ext_vector_type(16))) _Float16 v16h;
typedef __attribute__((ext_vector_type(8)))  float    v8f;

constexpr int   B       = 4;
constexpr int   P       = 1024;   // P1 == P2
constexpr int   D       = 256;
constexpr float EPS_R   = 0.1f;
constexpr float INV_EPS = 10.0f;
constexpr float THRESH  = 0.1f;
constexpr int   MAX_ITER = 20;

// ---- workspace layout (float offsets) ----
constexpr size_t OFF_C    = 0;                        // B*P*P
constexpr size_t OFF_CT   = (size_t)B * P * P;        // B*P*P (transpose)
constexpr size_t OFF_NX   = 2 * (size_t)B * P * P;    // B*P
constexpr size_t OFF_NY   = OFF_NX + (size_t)B * P;   // B*P
constexpr size_t OFF_U    = OFF_NY + (size_t)B * P;   // B*P
constexpr size_t OFF_V    = OFF_U  + (size_t)B * P;   // B*P
constexpr size_t OFF_ERR  = OFF_V  + (size_t)B * P;   // 1
constexpr size_t OFF_LOGQ = OFF_ERR + 1;              // 1
constexpr size_t OFF_DONE = OFF_LOGQ + 1;             // 1 (int flag)

__device__ __forceinline__ float wave_sum(float v) {
#pragma unroll
    for (int m = 16; m >= 1; m >>= 1) v += __shfl_xor(v, m, 32);
    return v;
}
__device__ __forceinline__ float wave_max(float v) {
#pragma unroll
    for (int m = 16; m >= 1; m >>= 1) v = fmaxf(v, __shfl_xor(v, m, 32));
    return v;
}

__device__ __forceinline__ v16h pack16(float4 a, float4 b, float4 c, float4 d) {
    v16h r;
    r[0]  = (_Float16)a.x; r[1]  = (_Float16)a.y; r[2]  = (_Float16)a.z; r[3]  = (_Float16)a.w;
    r[4]  = (_Float16)b.x; r[5]  = (_Float16)b.y; r[6]  = (_Float16)b.z; r[7]  = (_Float16)b.w;
    r[8]  = (_Float16)c.x; r[9]  = (_Float16)c.y; r[10] = (_Float16)c.z; r[11] = (_Float16)c.w;
    r[12] = (_Float16)d.x; r[13] = (_Float16)d.y; r[14] = (_Float16)d.z; r[15] = (_Float16)d.w;
    return r;
}

// ---- init: zero u, v, err, done, cost slots; compute log(q) ----
__global__ void sk_init(float* __restrict__ W, const float* __restrict__ q,
                        float* __restrict__ out) {
    int idx = blockIdx.x * blockDim.x + threadIdx.x;
    if (idx < B * P) {
        W[OFF_U + idx] = 0.0f;
        W[OFF_V + idx] = 0.0f;
    }
    if (idx == 0) {
        W[OFF_ERR] = 0.0f;
        ((int*)(W + OFF_DONE))[0] = 0;
        W[OFF_LOGQ] = logf(q[0]);
        out[0] = 0.0f; out[1] = 0.0f; out[2] = 0.0f; out[3] = 0.0f;
    }
}

// ---- row L2 norms of x and y: one wave32 per row of 256 floats ----
__global__ __launch_bounds__(256) void sk_norms(const float* __restrict__ x,
                                                const float* __restrict__ y,
                                                float* __restrict__ W) {
    int gw   = (blockIdx.x * blockDim.x + threadIdx.x) >> 5;  // global wave id
    int lane = threadIdx.x & 31;
    const float* src; float* dst; int r;
    if (gw < B * P) { src = x; dst = W + OFF_NX; r = gw; }
    else            { src = y; dst = W + OFF_NY; r = gw - B * P; }
    const float4* p = (const float4*)(src + (size_t)r * D);
    float4 a = p[lane * 2], b = p[lane * 2 + 1];
    float s = a.x*a.x + a.y*a.y + a.z*a.z + a.w*a.w
            + b.x*b.x + b.y*b.y + b.z*b.z + b.w*b.w;
    s = wave_sum(s);
    if (lane == 0) dst[r] = sqrtf(s);
}

// ---- cost GEMM: C = 1 - (x . y^T) / max(|x||y|, 1e-8), plus transpose ----
// One wave computes a 16(M) x 64(N) strip via 4 accumulators of
// v_wmma_f32_16x16x32_f16. NT orientation: A and B lane data are both
// contiguous runs from row-major x / y, no transpose loads needed.
__global__ __launch_bounds__(128) void sk_cost(const float* __restrict__ x,
                                               const float* __restrict__ y,
                                               float* __restrict__ W) {
    const int b    = blockIdx.z;
    const int wave = threadIdx.x >> 5;
    const int lane = threadIdx.x & 31;
    const int n    = lane & 15;    // M-index for A, N-index for B/D
    const int h    = lane >> 4;    // half-wave selector
    const int i0   = blockIdx.y * 64 + wave * 16;
    const int j0   = blockIdx.x * 64;

    v8f acc[4] = {v8f{}, v8f{}, v8f{}, v8f{}};

    const float* xrow = x + ((size_t)(b * P) + i0 + n) * D;

#pragma unroll
    for (int k0 = 0; k0 < D; k0 += 32) {
        // A: 16x32 f16. Lane (m + 16h): K = 16h + 0..15 (contiguous).
        const float4* alo = (const float4*)(xrow + k0 + 16 * h);
        const float4* ahi = (const float4*)(xrow + k0 + 16 * h + 8);
        v16h a = pack16(alo[0], alo[1], ahi[0], ahi[1]);

#pragma unroll
        for (int t = 0; t < 4; t++) {
            // B: 32x16 f16, B[k][n] = y[j0+16t+n][k].
            // Lane (n + 16h): K = 16h + 0..15 (contiguous in y's row).
            const float4* bp = (const float4*)(y +
                ((size_t)(b * P) + j0 + 16 * t + n) * D + k0 + 16 * h);
            v16h bb = pack16(bp[0], bp[1], bp[2], bp[3]);
            acc[t] = __builtin_amdgcn_wmma_f32_16x16x32_f16(
                false, a, false, bb, (short)0, acc[t], false, false);
        }
    }

    // Epilogue: D layout lane (n+16h), VGPR r -> (M = r+8h, N = n).
    float nxv[8];
#pragma unroll
    for (int r = 0; r < 8; r++)
        nxv[r] = W[OFF_NX + b * P + i0 + 8 * h + r];

#pragma unroll
    for (int t = 0; t < 4; t++) {
        const int j   = j0 + 16 * t + n;
        const float ny = W[OFF_NY + b * P + j];
#pragma unroll
        for (int r = 0; r < 8; r++) {
            const int i   = i0 + 8 * h + r;
            const float d = acc[t][r];
            const float c = 1.0f - d / fmaxf(nxv[r] * ny, 1e-8f);
            W[OFF_C  + ((size_t)(b * P) + i) * P + j] = c;
            W[OFF_CT + ((size_t)(b * P) + j) * P + i] = c;
        }
    }
}

// ---- u update: one wave per (b,i) row; single-pass register LSE ----
// u_new = eps * (log_mu - LSE_j((v_j - C_ij - logq)/eps))   (u cancels)
__global__ __launch_bounds__(256) void sk_u(float* __restrict__ W) {
    if (((const volatile int*)(W + OFF_DONE))[0]) return;
    const int gw   = (blockIdx.x * blockDim.x + threadIdx.x) >> 5;
    const int lane = threadIdx.x & 31;
    const int b = gw >> 10, i = gw & (P - 1);
    const float* Crow = W + OFF_C + ((size_t)(b * P) + i) * P;
    const float* vrow = W + OFF_V + b * P;
    const float logq  = W[OFF_LOGQ];

    float tv[32];
    float m = -1e30f;
#pragma unroll
    for (int t = 0; t < 32; t++) {
        const int j = lane + 32 * t;
        const float val = (vrow[j] - Crow[j] - logq) * INV_EPS;
        tv[t] = val;
        m = fmaxf(m, val);
    }
    m = wave_max(m);
    float s = 0.0f;
#pragma unroll
    for (int t = 0; t < 32; t++) s += expf(tv[t] - m);
    s = wave_sum(s);

    if (lane == 0) {
        const float lse    = m + logf(s);
        const float log_mu = logf(1.0f / (float)P + 1e-8f);
        const float un     = EPS_R * (log_mu - lse);
        float* up = W + OFF_U + b * P + i;
        atomicAdd(W + OFF_ERR, fabsf(un - *up));
        *up = un;
    }
}

// ---- v update: same over C^T using fresh u ----
__global__ __launch_bounds__(256) void sk_v(float* __restrict__ W) {
    if (((const volatile int*)(W + OFF_DONE))[0]) return;
    const int gw   = (blockIdx.x * blockDim.x + threadIdx.x) >> 5;
    const int lane = threadIdx.x & 31;
    const int b = gw >> 10, j = gw & (P - 1);
    const float* CTrow = W + OFF_CT + ((size_t)(b * P) + j) * P;
    const float* urow  = W + OFF_U + b * P;
    const float logq   = W[OFF_LOGQ];

    float tv[32];
    float m = -1e30f;
#pragma unroll
    for (int t = 0; t < 32; t++) {
        const int i = lane + 32 * t;
        const float val = (urow[i] - CTrow[i] - logq) * INV_EPS;
        tv[t] = val;
        m = fmaxf(m, val);
    }
    m = wave_max(m);
    float s = 0.0f;
#pragma unroll
    for (int t = 0; t < 32; t++) s += expf(tv[t] - m);
    s = wave_sum(s);

    if (lane == 0) {
        const float lse    = m + logf(s);
        const float log_nu = logf(1.0f / (float)P + 1e-8f);
        W[OFF_V + b * P + j] = EPS_R * (log_nu - lse);
    }
}

// ---- early-exit flag: done |= (err/B < THRESH); reset err ----
__global__ void sk_flag(float* __restrict__ W) {
    if (W[OFF_ERR] * (1.0f / (float)B) < THRESH)
        ((int*)(W + OFF_DONE))[0] = 1;
    W[OFF_ERR] = 0.0f;
}

// ---- final: pi = exp(M(u,v)); cost_b = sum(pi * C) ----
__global__ __launch_bounds__(256) void sk_final(const float* __restrict__ W,
                                                float* __restrict__ out) {
    const int gw   = (blockIdx.x * blockDim.x + threadIdx.x) >> 5;
    const int lane = threadIdx.x & 31;
    const int b = gw >> 10, i = gw & (P - 1);
    const float* Crow = W + OFF_C + ((size_t)(b * P) + i) * P;
    const float ui    = W[OFF_U + b * P + i];
    const float* vrow = W + OFF_V + b * P;
    const float logq  = W[OFF_LOGQ];
    float* prow = out + 4 + ((size_t)(b * P) + i) * P;

    float acc = 0.0f;
#pragma unroll 8
    for (int t = 0; t < 32; t++) {
        const int j = lane + 32 * t;
        const float c  = Crow[j];
        const float pi = expf((ui + vrow[j] - c - logq) * INV_EPS);
        prow[j] = pi;
        acc += pi * c;
    }
    acc = wave_sum(acc);
    if (lane == 0) atomicAdd(out + b, acc);
}

extern "C" void kernel_launch(void* const* d_in, const int* in_sizes, int n_in,
                              void* d_out, int out_size, void* d_ws, size_t ws_size,
                              hipStream_t stream) {
    const float* x = (const float*)d_in[0];   // (4,1024,256) fp32
    const float* y = (const float*)d_in[1];   // (4,1024,256) fp32
    const float* q = (const float*)d_in[2];   // scalar
    float* out = (float*)d_out;               // [cost(4) | pi(4*1024*1024)]
    float* W   = (float*)d_ws;

    sk_init <<<(B * P + 255) / 256, 256, 0, stream>>>(W, q, out);
    sk_norms<<<(2 * B * P) / 8,     256, 0, stream>>>(x, y, W);
    sk_cost <<<dim3(P / 64, P / 64, B), 128, 0, stream>>>(x, y, W);
    for (int it = 0; it < MAX_ITER; ++it) {
        sk_u   <<<(B * P) / 8, 256, 0, stream>>>(W);
        sk_v   <<<(B * P) / 8, 256, 0, stream>>>(W);
        sk_flag<<<1, 1, 0, stream>>>(W);
    }
    sk_final<<<(B * P) / 8, 256, 0, stream>>>(W, out);
}
